// GATv5_26414048870729
// MI455X (gfx1250) — matrix operand order repaired and checked by reading, not compiled, not run on hardware
//
#include <hip/hip_runtime.h>
#include <hip/hip_bf16.h>

typedef __attribute__((ext_vector_type(16))) __bf16 v16bf;
typedef __attribute__((ext_vector_type(8)))  float  v8f;

#define NNODES   10000
#define K_IN     9998
#define NEDGE    320000
#define ETOT     (NEDGE + NNODES)
#define NB       80            // combined B columns: 8 (Wl1) + 8 (Wr1) + 64 (enc_W1)
#define KT       313           // ceil(9998/32)
#define MTILES   (NNODES/16)   // 625
#define NEG_SLOPE 0.2f
#define ENC_NEG_MAX 0x00800000u  // monotone-encoded -FLT_MAX

// ---------- prep: pre-swizzle combined B into WMMA bf16 fragment layout ----------
// Fragment (t, nt): 32 lanes x 16 bf16. Lane L: col = nt*16 + (L&15),
// K-range = t*32 + (L<16 ? 0..15 : 16..31) (element e -> K = base+e).
__global__ void k_prep(const float* __restrict__ Wl1, const float* __restrict__ Wr1,
                       const float* __restrict__ encW1,
                       const float* __restrict__ bl1, const float* __restrict__ br1,
                       const float* __restrict__ encb1,
                       __bf16* __restrict__ bswz, float* __restrict__ biascomb) {
  int idx = blockIdx.x * blockDim.x + threadIdx.x;
  if (idx < NB) {
    biascomb[idx] = (idx < 8) ? bl1[idx] : (idx < 16) ? br1[idx - 8] : encb1[idx - 16];
  }
  if (idx >= KT * 5 * 32 * 16) return;
  int e = idx & 15;
  int L = (idx >> 4) & 31;
  int n = (idx >> 9) % 5;
  int t = idx / (5 * 32 * 16);
  int c  = L & 15;
  int kb = (L < 16) ? 0 : 16;
  int k  = t * 32 + kb + e;
  int col = n * 16 + c;
  float v = 0.0f;
  if (k < K_IN) {
    v = (col < 8)  ? Wl1[k * 8 + col]
      : (col < 16) ? Wr1[k * 8 + (col - 8)]
                   : encW1[k * 64 + (col - 16)];
  }
  bswz[idx] = (__bf16)v;
}

// ---------- init segment-softmax state for both layers ----------
__global__ void k_init(unsigned* __restrict__ m1, unsigned* __restrict__ m2,
                       float* __restrict__ den1, float* __restrict__ den2,
                       float* __restrict__ acc1, float* __restrict__ acc2) {
  int i = blockIdx.x * blockDim.x + threadIdx.x;
  if (i < 2 * NNODES) { m1[i] = ENC_NEG_MAX; m2[i] = ENC_NEG_MAX; den1[i] = 0.f; den2[i] = 0.f; }
  if (i < 8 * NNODES) { acc1[i] = 0.f; acc2[i] = 0.f; }
}

// ---------- fused WMMA GEMM: out[10000,80] = x @ [Wl1|Wr1|encW1[:9998]] + biascomb ----------
__global__ __launch_bounds__(256) void k_gemm(const float* __restrict__ x,
                                              const __bf16* __restrict__ bswz,
                                              const float* __restrict__ biascomb,
                                              float* __restrict__ outbig) {
  const int lane  = threadIdx.x & 31;
  const int wave  = threadIdx.x >> 5;
  const int mtile = blockIdx.x * 8 + wave;
  if (mtile >= MTILES) return;

  const int  row  = lane & 15;
  const long m    = (long)mtile * 16 + row;
  const float* __restrict__ xrow = x + m * (long)K_IN;
  const int kb0 = (lane < 16) ? 0 : 16;  // A layout: lanes<16 hold K 0..15; lanes>=16 hold 16..31
  const int kb1 = kb0 + 8;

  v8f acc[5] = {v8f{}, v8f{}, v8f{}, v8f{}, v8f{}};
  union AU { v16bf v; __bf16 e[16]; };

  for (int t = 0; t < KT - 1; ++t) {
    const int k0 = t * 32;
    __builtin_prefetch(xrow + k0 + 32 + kb0, 0, 0);   // global_prefetch next K-tile
    AU a;
    const float2* p0 = reinterpret_cast<const float2*>(xrow + k0 + kb0);
    const float2* p1 = reinterpret_cast<const float2*>(xrow + k0 + kb1);
#pragma unroll
    for (int i = 0; i < 4; ++i) {
      float2 f0 = p0[i], f1 = p1[i];
      a.e[2 * i]     = (__bf16)f0.x;  a.e[2 * i + 1]     = (__bf16)f0.y;
      a.e[8 + 2 * i] = (__bf16)f1.x;  a.e[8 + 2 * i + 1] = (__bf16)f1.y;
    }
    const v16bf* bp = reinterpret_cast<const v16bf*>(bswz + ((long)t * 5 * 32 + lane) * 16);
#pragma unroll
    for (int nt = 0; nt < 5; ++nt) {
      acc[nt] = __builtin_amdgcn_wmma_f32_16x16x32_bf16(
          false, a.v, false, bp[nt * 32], (short)0, acc[nt], false, false);
    }
  }
  { // tail tile: K = 9984..10015, mask K >= 9998 (B already zero-padded)
    const int k0 = (KT - 1) * 32;
    AU a;
#pragma unroll
    for (int i = 0; i < 8; ++i) {
      int ka = k0 + kb0 + i, kc = k0 + kb1 + i;
      a.e[i]     = (__bf16)((ka < K_IN) ? xrow[ka] : 0.0f);
      a.e[8 + i] = (__bf16)((kc < K_IN) ? xrow[kc] : 0.0f);
    }
    const v16bf* bp = reinterpret_cast<const v16bf*>(bswz + ((long)(KT - 1) * 5 * 32 + lane) * 16);
#pragma unroll
    for (int nt = 0; nt < 5; ++nt) {
      acc[nt] = __builtin_amdgcn_wmma_f32_16x16x32_bf16(
          false, a.v, false, bp[nt * 32], (short)0, acc[nt], false, false);
    }
  }
  // C/D layout: lanes<16 -> M=i, N=lane; lanes>=16 -> M=8+i, N=lane-16
  const int colb = lane & 15;
  const int rowo = (lane < 16) ? 0 : 8;
#pragma unroll
  for (int nt = 0; nt < 5; ++nt) {
#pragma unroll
    for (int i = 0; i < 8; ++i) {
      int  col = nt * 16 + colb;
      long mm  = (long)mtile * 16 + rowo + i;
      outbig[mm * NB + col] = acc[nt][i] + biascomb[col];
    }
  }
}

// ---------- edge pass A: logits + segment max (monotone-uint atomicMax) ----------
__global__ void k_edge_logits(const int* __restrict__ esrc, const int* __restrict__ edst,
                              const float* __restrict__ xl, int ldl,
                              const float* __restrict__ xr, int ldr,
                              const float* __restrict__ att,
                              float* __restrict__ lg, unsigned* __restrict__ mmax) {
  int e = blockIdx.x * blockDim.x + threadIdx.x;
  if (e >= ETOT) return;
  int s = (e < NEDGE) ? esrc[e] : (e - NEDGE);   // add_self_loops
  int d = (e < NEDGE) ? edst[e] : (e - NEDGE);
  const float* pl = xl + (long)s * ldl;
  const float* pr = xr + (long)d * ldr;
#pragma unroll
  for (int h = 0; h < 2; ++h) {
    float l = 0.0f;
#pragma unroll
    for (int c = 0; c < 4; ++c) {
      float v = pl[h * 4 + c] + pr[h * 4 + c];
      v = (v > 0.0f) ? v : NEG_SLOPE * v;
      l += att[h * 4 + c] * v;
    }
    lg[(long)e * 2 + h] = l;
    unsigned u = __float_as_uint(l);
    u ^= (u >> 31) ? 0xFFFFFFFFu : 0x80000000u;
    atomicMax(&mmax[d * 2 + h], u);
  }
}

// ---------- edge pass B: ea = exp(logit - max); denom += ea ----------
__global__ void k_edge_exp(const int* __restrict__ edst,
                           float* __restrict__ lg, const unsigned* __restrict__ mmax,
                           float* __restrict__ den) {
  int e = blockIdx.x * blockDim.x + threadIdx.x;
  if (e >= ETOT) return;
  int d = (e < NEDGE) ? edst[e] : (e - NEDGE);
#pragma unroll
  for (int h = 0; h < 2; ++h) {
    unsigned u = mmax[d * 2 + h];
    float mv = (u & 0x80000000u) ? __uint_as_float(u ^ 0x80000000u) : __uint_as_float(~u);
    float ea = expf(lg[(long)e * 2 + h] - mv);
    lg[(long)e * 2 + h] = ea;
    atomicAdd(&den[d * 2 + h], ea);
  }
}

// ---------- edge pass C: acc[dst] += xl[src] * alpha ----------
__global__ void k_edge_agg(const int* __restrict__ esrc, const int* __restrict__ edst,
                           const float* __restrict__ xl, int ldl,
                           const float* __restrict__ lg, const float* __restrict__ den,
                           float* __restrict__ acc) {
  int e = blockIdx.x * blockDim.x + threadIdx.x;
  if (e >= ETOT) return;
  int s = (e < NEDGE) ? esrc[e] : (e - NEDGE);
  int d = (e < NEDGE) ? edst[e] : (e - NEDGE);
  const float* pl = xl + (long)s * ldl;
#pragma unroll
  for (int h = 0; h < 2; ++h) {
    float alpha = lg[(long)e * 2 + h] / (den[d * 2 + h] + 1e-16f);
#pragma unroll
    for (int c = 0; c < 4; ++c) {
      atomicAdd(&acc[(long)d * 8 + h * 4 + c], pl[h * 4 + c] * alpha);
    }
  }
}

// ---------- node pass 1: x1 = relu(gat1+bias1)@lin1 + b; rank-1 layer-2 transforms ----------
__global__ void k_node1(const float* __restrict__ acc1, const float* __restrict__ bias1,
                        const float* __restrict__ lin1W, const float* __restrict__ lin1b,
                        const float* __restrict__ Wl2, const float* __restrict__ bl2,
                        const float* __restrict__ Wr2, const float* __restrict__ br2,
                        float* __restrict__ x1, float* __restrict__ xl2, float* __restrict__ xr2) {
  int v = blockIdx.x * blockDim.x + threadIdx.x;
  if (v >= NNODES) return;
  float s = lin1b[0];
#pragma unroll
  for (int j = 0; j < 8; ++j) {
    float g = fmaxf(acc1[(long)v * 8 + j] + bias1[j], 0.0f);
    s += g * lin1W[j];
  }
  x1[v] = s;
#pragma unroll
  for (int j = 0; j < 8; ++j) {
    xl2[(long)v * 8 + j] = s * Wl2[j] + bl2[j];
    xr2[(long)v * 8 + j] = s * Wr2[j] + br2[j];
  }
}

// ---------- node pass 2: x2, rank-1 encoder corrections, 64->32->1 MLP ----------
__global__ void k_node2(const float* __restrict__ acc2, const float* __restrict__ bias2,
                        const float* __restrict__ lin2W, const float* __restrict__ lin2b,
                        const float* __restrict__ outbig, const float* __restrict__ x1,
                        const float* __restrict__ encW1, const float* __restrict__ encW2,
                        const float* __restrict__ encb2, const float* __restrict__ encW3,
                        const float* __restrict__ encb3, float* __restrict__ out) {
  int v = blockIdx.x * blockDim.x + threadIdx.x;
  if (v >= NNODES) return;
  float s = lin2b[0];
#pragma unroll
  for (int j = 0; j < 8; ++j) {
    float g = fmaxf(acc2[(long)v * 8 + j] + bias2[j], 0.0f);
    s += g * lin2W[j];
  }
  float x2v = s;
  float x1v = x1[v];
  const float* hp = outbig + (long)v * NB + 16;   // enc partial (bias already fused)
  float h32[32];
#pragma unroll
  for (int i = 0; i < 32; ++i) h32[i] = encb2[i];
  for (int j = 0; j < 64; ++j) {
    float h64j = fmaxf(hp[j] + x1v * encW1[9998 * 64 + j] + x2v * encW1[9999 * 64 + j], 0.0f);
#pragma unroll
    for (int i = 0; i < 32; ++i) h32[i] += h64j * encW2[j * 32 + i];
  }
  float o = encb3[0];
#pragma unroll
  for (int i = 0; i < 32; ++i) o += fmaxf(h32[i], 0.0f) * encW3[i];
  out[v] = o;
}

extern "C" void kernel_launch(void* const* d_in, const int* in_sizes, int n_in,
                              void* d_out, int out_size, void* d_ws, size_t ws_size,
                              hipStream_t stream) {
  const float* x     = (const float*)d_in[0];
  const int*   ei    = (const int*)d_in[1];
  const float* Wl1   = (const float*)d_in[2];
  const float* bl1   = (const float*)d_in[3];
  const float* Wr1   = (const float*)d_in[4];
  const float* br1   = (const float*)d_in[5];
  const float* att1  = (const float*)d_in[6];
  const float* bias1 = (const float*)d_in[7];
  const float* lin1W = (const float*)d_in[8];
  const float* lin1b = (const float*)d_in[9];
  const float* Wl2   = (const float*)d_in[10];
  const float* bl2   = (const float*)d_in[11];
  const float* Wr2   = (const float*)d_in[12];
  const float* br2   = (const float*)d_in[13];
  const float* att2  = (const float*)d_in[14];
  const float* bias2 = (const float*)d_in[15];
  const float* lin2W = (const float*)d_in[16];
  const float* lin2b = (const float*)d_in[17];
  const float* encW1 = (const float*)d_in[18];
  const float* encb1 = (const float*)d_in[19];
  const float* encW2 = (const float*)d_in[20];
  const float* encb2 = (const float*)d_in[21];
  const float* encW3 = (const float*)d_in[22];
  const float* encb3 = (const float*)d_in[23];

  const int* esrc = ei;
  const int* edst = ei + NEDGE;

  // workspace bump allocator (256B aligned)
  char* p = (char*)d_ws;
  auto balloc = [&](size_t bytes) -> char* {
    char* r = p;
    p += (bytes + 255) & ~(size_t)255;
    return r;
  };
  __bf16*   bswz     = (__bf16*)  balloc((size_t)KT * 5 * 32 * 16 * sizeof(__bf16));
  float*    biascomb = (float*)   balloc(NB * sizeof(float));
  float*    outbig   = (float*)   balloc((size_t)NNODES * NB * sizeof(float));
  unsigned* m1       = (unsigned*)balloc((size_t)2 * NNODES * sizeof(unsigned));
  unsigned* m2       = (unsigned*)balloc((size_t)2 * NNODES * sizeof(unsigned));
  float*    den1     = (float*)   balloc((size_t)2 * NNODES * sizeof(float));
  float*    den2     = (float*)   balloc((size_t)2 * NNODES * sizeof(float));
  float*    acc1     = (float*)   balloc((size_t)8 * NNODES * sizeof(float));
  float*    acc2     = (float*)   balloc((size_t)8 * NNODES * sizeof(float));
  float*    lg1      = (float*)   balloc((size_t)2 * ETOT * sizeof(float));
  float*    lg2      = (float*)   balloc((size_t)2 * ETOT * sizeof(float));
  float*    x1       = (float*)   balloc((size_t)NNODES * sizeof(float));
  float*    xl2      = (float*)   balloc((size_t)8 * NNODES * sizeof(float));
  float*    xr2      = (float*)   balloc((size_t)8 * NNODES * sizeof(float));

  const int eg = (ETOT + 255) / 256;
  const int ng = (NNODES + 255) / 256;

  k_prep<<<(KT * 5 * 32 * 16 + 255) / 256, 256, 0, stream>>>(Wl1, Wr1, encW1, bl1, br1, encb1,
                                                             bswz, biascomb);
  k_init<<<(8 * NNODES + 255) / 256, 256, 0, stream>>>(m1, m2, den1, den2, acc1, acc2);
  k_gemm<<<(MTILES + 7) / 8, 256, 0, stream>>>(x, bswz, biascomb, outbig);

  // GATv2 layer 1: xl = outbig[:, :8], xr = outbig[:, 8:16] (ld = 80, biases fused)
  k_edge_logits<<<eg, 256, 0, stream>>>(esrc, edst, outbig, NB, outbig + 8, NB, att1, lg1, m1);
  k_edge_exp   <<<eg, 256, 0, stream>>>(edst, lg1, m1, den1);
  k_edge_agg   <<<eg, 256, 0, stream>>>(esrc, edst, outbig, NB, lg1, den1, acc1);
  k_node1      <<<ng, 256, 0, stream>>>(acc1, bias1, lin1W, lin1b, Wl2, bl2, Wr2, br2,
                                        x1, xl2, xr2);
  // GATv2 layer 2
  k_edge_logits<<<eg, 256, 0, stream>>>(esrc, edst, xl2, 8, xr2, 8, att2, lg2, m2);
  k_edge_exp   <<<eg, 256, 0, stream>>>(edst, lg2, m2, den2);
  k_edge_agg   <<<eg, 256, 0, stream>>>(esrc, edst, xl2, 8, lg2, den2, acc2);
  k_node2      <<<ng, 256, 0, stream>>>(acc2, bias2, lin2W, lin2b, outbig, x1,
                                        encW1, encW2, encb2, encW3, encb3, (float*)d_out);
}
